// LinearAttention_15985868276494
// MI455X (gfx1250) — compile-verified, matrix operand order checked by reading, not run on hardware
//
#include <hip/hip_runtime.h>
#include <math.h>

// Performer linear attention, fused for MI455X (gfx1250, wave32, WMMA).
// n=2, l=2048, h=8, e=64, m=2048.

#define Ln 2048
#define Hh 8
#define Ee 64
#define Mm 2048
#define NHt (2 * Hh)

#define NORMK 0.35355339059327373f   /* 64^-0.25 */
#define RATIO 0.022097086912079608f  /* 2048^-0.5 */
#define RKEPS (RATIO * 1.0e-4f)      /* ratio * kernel eps */

typedef __attribute__((ext_vector_type(16))) __bf16 bf16x16;
typedef __attribute__((ext_vector_type(8)))  float  floatx8;

__device__ __forceinline__ floatx8 zero8() {
  floatx8 z = {0.f, 0.f, 0.f, 0.f, 0.f, 0.f, 0.f, 0.f};
  return z;
}

__device__ __forceinline__ floatx8 wmma_bf16(bf16x16 a, bf16x16 b, floatx8 c) {
  return __builtin_amdgcn_wmma_f32_16x16x32_bf16(
      false, a, false, b, (short)0, c, false, false);
}

// 16-bit A/B fragment K mapping (ISA 7.12.2): VGPR v holds K pair
// {k, k+1}; v<4 -> k=2v, v>=4 -> k=16+2(v-4); upper lane half adds 8.
__device__ __forceinline__ int kmap_lo(int v) {
  return (v < 4) ? (2 * v) : (16 + 2 * (v - 4));
}

// Fragment whose lane index selects a row of a row-major matrix and K runs
// contiguously along that row (A tiles from LDS).
template <typename SrcT>
__device__ __forceinline__ bf16x16 frag_row(const SrcT* base, int ld, int roff,
                                            int k0, int lane) {
  bf16x16 f;
  const int r = roff + (lane & 15);
  const int kh = (lane >> 4) * 8;
#pragma unroll
  for (int v = 0; v < 8; ++v) {
    const int k = k0 + kmap_lo(v) + kh;
    f[2 * v]     = (__bf16)base[(size_t)r * ld + k];
    f[2 * v + 1] = (__bf16)base[(size_t)r * ld + k + 1];
  }
  return f;
}

// Fragment whose lane index selects a column; K strides by ld (LDS tiles).
template <typename SrcT>
__device__ __forceinline__ bf16x16 frag_col(const SrcT* base, int ld, int koff,
                                            int c0, int lane) {
  bf16x16 f;
  const int c = c0 + (lane & 15);
  const int kh = (lane >> 4) * 8;
#pragma unroll
  for (int v = 0; v < 8; ++v) {
    const int k = koff + kmap_lo(v) + kh;
    f[2 * v]     = (__bf16)base[(size_t)k * ld + c];
    f[2 * v + 1] = (__bf16)base[(size_t)(k + 1) * ld + c];
  }
  return f;
}

// Fragment-ready packed layout: each lane's 16 bf16 values are contiguous
// (32 bytes) -> one 32B vector load (2x global_load_b128), coalesced.
__device__ __forceinline__ bf16x16 frag_packed(const __bf16* base, int idx,
                                               int lane) {
  return *(const bf16x16*)(base + ((size_t)idx * 32 + lane) * 16);
}

// -------- Kernel 0: pre-swizzle projection P into frag-ready bf16 ----------
// Ps[tile=128][kidx=2][lane=32][16 bf16]; 8192 threads total.
__global__ __launch_bounds__(256) void k_prep_p(const float* __restrict__ P,
                                                __bf16* __restrict__ Ps) {
  const int id = blockIdx.x * 256 + threadIdx.x;  // 0..8191
  const int mt = id >> 6;
  const int kidx = (id >> 5) & 1;
  const int lane = id & 31;
  const int mrow = mt * 16 + (lane & 15);
  const int kh = (lane >> 4) * 8;
  __bf16* dst = Ps + ((size_t)(mt * 2 + kidx) * 32 + lane) * 16;
#pragma unroll
  for (int v = 0; v < 8; ++v) {
    const int k = kidx * 32 + kmap_lo(v) + kh;
    dst[2 * v]     = (__bf16)P[(size_t)mrow * 64 + k];
    dst[2 * v + 1] = (__bf16)P[(size_t)mrow * 64 + k + 1];
  }
}

// -------- Kernel 1: per-row shift = 0.5*||x*norm||^2 + max_m(data_dash) ----
// grid 256 = (n*h)*(L/128); block 256 = 8 waves, wave w owns 16 rows.
__global__ __launch_bounds__(256) void k_rowshift(
    const float* __restrict__ X, const __bf16* __restrict__ Ps,
    float* __restrict__ shift) {
  __shared__ __bf16 At[128 * 64];
  __shared__ float ssq[128];
  const int bid = blockIdx.x;
  const int lblk = bid & 15;
  const int nh = bid >> 4;
  const int n_i = nh / Hh, h_i = nh % Hh;
  const int l0 = lblk * 128;
  const int tid = threadIdx.x, lane = tid & 31, wave = tid >> 5;

  if (tid < 128) ssq[tid] = 0.f;
  __syncthreads();
  {
    const int row = tid >> 1, seg = tid & 1;
    const float* src =
        X + ((size_t)((n_i * Ln + l0 + row) * Hh + h_i)) * Ee + seg * 32;
    float s = 0.f;
#pragma unroll
    for (int c = 0; c < 32; ++c) {
      const float x = src[c] * NORMK;
      At[row * 64 + seg * 32 + c] = (__bf16)x;
      s += x * x;
    }
    atomicAdd(&ssq[row], s);
  }
  __syncthreads();

  const int r0 = wave * 16;
  const bf16x16 a0 = frag_row(At, 64, r0, 0, lane);
  const bf16x16 a1 = frag_row(At, 64, r0, 32, lane);
  float mx[8];
#pragma unroll
  for (int r = 0; r < 8; ++r) mx[r] = -3.0e38f;

  for (int mt = 0; mt < Mm / 16; ++mt) {
    const bf16x16 b0 = frag_packed(Ps, mt * 2, lane);
    const bf16x16 b1 = frag_packed(Ps, mt * 2 + 1, lane);
    floatx8 c = zero8();
    c = wmma_bf16(a0, b0, c);
    c = wmma_bf16(a1, b1, c);
#pragma unroll
    for (int r = 0; r < 8; ++r) mx[r] = fmaxf(mx[r], c[r]);
  }
  // reduce across the 16 lanes of each half (xor<=8 keeps lane half fixed)
#pragma unroll
  for (int d = 1; d < 16; d <<= 1)
#pragma unroll
    for (int r = 0; r < 8; ++r) mx[r] = fmaxf(mx[r], __shfl_xor(mx[r], d, 32));

  if ((lane & 15) == 0) {
    const int half = lane >> 4;
#pragma unroll
    for (int r = 0; r < 8; ++r) {
      const int row = r0 + r + 8 * half;
      shift[(size_t)nh * Ln + l0 + row] = 0.5f * ssq[row] + mx[r];
    }
  }
}

// -------- Kernel 2: KV[e,m] = sum_s Kf[s,m] V[s,e]; ksum[m] = sum_s Kf -----
// grid 256 = (n*h)*(M/128); block 256 = 8 waves, wave w owns m-cols w*16..+16.
// KV is written out pre-swizzled (frag-ready) for k_out's B operands.
__global__ __launch_bounds__(256) void k_kv(
    const float* __restrict__ K, const float* __restrict__ V,
    const __bf16* __restrict__ Ps, const float* __restrict__ shiftK,
    float* __restrict__ ksum, __bf16* __restrict__ KVs) {
  __shared__ __bf16 Kt[32 * 64];
  __shared__ __bf16 Vt[32 * 64];
  __shared__ __bf16 Kf[32 * 128];
  __shared__ __bf16 KVt[64 * 128];
  __shared__ float sh[32];
  const int bid = blockIdx.x;
  const int mblk = bid & 15;
  const int nh = bid >> 4;
  const int n_i = nh / Hh, h_i = nh % Hh;
  const int tid = threadIdx.x, lane = tid & 31, wave = tid >> 5;
  const int half = lane >> 4;

  // B = P^T fragments for this wave's m slice (loop invariant, packed)
  const bf16x16 bp0 = frag_packed(Ps, (mblk * 8 + wave) * 2, lane);
  const bf16x16 bp1 = frag_packed(Ps, (mblk * 8 + wave) * 2 + 1, lane);

  floatx8 acc[4];
#pragma unroll
  for (int t = 0; t < 4; ++t) acc[t] = zero8();
  float ksacc = 0.f;

  for (int s0 = 0; s0 < Ln; s0 += 32) {
    __syncthreads();
    {
      const int row = tid >> 3, seg = tid & 7;
      const size_t base =
          ((size_t)((n_i * Ln + s0 + row) * Hh + h_i)) * Ee + seg * 8;
#pragma unroll
      for (int c = 0; c < 8; ++c) {
        Kt[row * 64 + seg * 8 + c] = (__bf16)(K[base + c] * NORMK);
        Vt[row * 64 + seg * 8 + c] = (__bf16)(V[base + c]);
      }
      if (tid < 32) sh[tid] = shiftK[(size_t)nh * Ln + s0 + tid];
    }
    __syncthreads();

    // data_dash for 32 s-rows x this wave's 16 m-cols, then exp -> Kf (LDS)
    const bf16x16 ak0 = frag_row(Kt, 64, 0, 0, lane);
    const bf16x16 ak1 = frag_row(Kt, 64, 0, 32, lane);
    const bf16x16 ak2 = frag_row(Kt, 64, 16, 0, lane);
    const bf16x16 ak3 = frag_row(Kt, 64, 16, 32, lane);
    floatx8 c0 = zero8();
    c0 = wmma_bf16(ak0, bp0, c0);
    c0 = wmma_bf16(ak1, bp1, c0);
    floatx8 c1 = zero8();
    c1 = wmma_bf16(ak2, bp0, c1);
    c1 = wmma_bf16(ak3, bp1, c1);
    const int col = wave * 16 + (lane & 15);
#pragma unroll
    for (int r = 0; r < 8; ++r) {
      const int row = r + 8 * half;
      Kf[row * 128 + col] =
          (__bf16)(RATIO * __expf(c0[r] - sh[row]) + RKEPS);
      Kf[(16 + row) * 128 + col] =
          (__bf16)(RATIO * __expf(c1[r] - sh[16 + row]) + RKEPS);
    }
    __syncthreads();

    // KV_tile += V^T[64 x 32s] * Kf[32s x 16m]
    const bf16x16 bkf = frag_col(Kf, 128, 0, wave * 16, lane);
#pragma unroll
    for (int t = 0; t < 4; ++t) {
      const bf16x16 av = frag_col(Vt, 64, 0, t * 16, lane);
      acc[t] = wmma_bf16(av, bkf, acc[t]);
    }
    if (tid < 128) {
#pragma unroll
      for (int ss = 0; ss < 32; ++ss) ksacc += (float)Kf[ss * 128 + tid];
    }
  }

  // Stage KV tile through LDS, then emit in frag-ready packed layout:
  // KVs[nh][mchunk=64][t=4][lane=32][16 bf16]
  __syncthreads();
  {
    const int col = wave * 16 + (lane & 15);
#pragma unroll
    for (int t = 0; t < 4; ++t)
#pragma unroll
      for (int r = 0; r < 8; ++r) {
        const int e = t * 16 + r + 8 * half;
        KVt[e * 128 + col] = (__bf16)acc[t][r];
      }
  }
  __syncthreads();
#pragma unroll
  for (int i = 0; i < 2; ++i) {
    const int p = wave * 2 + i;       // 16 (mchunk_local, t) pairs per block
    const int mc = p >> 2, t = p & 3;
    const int e = t * 16 + (lane & 15);
    const int kh = (lane >> 4) * 8;
    bf16x16 f;
#pragma unroll
    for (int v = 0; v < 8; ++v) {
      const int k = mc * 32 + kmap_lo(v) + kh;
      f[2 * v]     = KVt[e * 128 + k];
      f[2 * v + 1] = KVt[e * 128 + k + 1];
    }
    const int idx = ((nh * 64) + (mblk * 4 + mc)) * 4 + t;
    *(bf16x16*)(KVs + ((size_t)idx * 32 + lane) * 16) = f;
  }
  if (tid < 128) ksum[(size_t)nh * Mm + mblk * 128 + tid] = ksacc;
}

// -------- Kernel 3: out[l,e] = z[l] * sum_m Qf[l,m] KV[e,m] ----------------
// grid 256 = (n*h)*(L/128); block 256 = 8 waves, wave w owns 16 l-rows.
__global__ __launch_bounds__(256) void k_out(
    const float* __restrict__ Q, const __bf16* __restrict__ Ps,
    const float* __restrict__ shiftQ, const float* __restrict__ ksum,
    const __bf16* __restrict__ KVs, float* __restrict__ out) {
  __shared__ __bf16 Qt[128 * 64];
  __shared__ float sh[128];
  __shared__ __bf16 Qf[8][16 * 32];  // per-wave Qf tile (wave-private)
  const int bid = blockIdx.x;
  const int lblk = bid & 15;
  const int nh = bid >> 4;
  const int n_i = nh / Hh, h_i = nh % Hh;
  const int l0 = lblk * 128;
  const int tid = threadIdx.x, lane = tid & 31, wave = tid >> 5;
  const int half = lane >> 4;

  {
    const int row = tid >> 1, seg = tid & 1;
    const size_t base =
        ((size_t)((n_i * Ln + l0 + row) * Hh + h_i)) * Ee + seg * 32;
#pragma unroll
    for (int c = 0; c < 32; ++c)
      Qt[row * 64 + seg * 32 + c] = (__bf16)(Q[base + c] * NORMK);
    if (tid < 128) sh[tid] = shiftQ[(size_t)nh * Ln + l0 + tid];
  }
  __syncthreads();

  const bf16x16 aq0 = frag_row(Qt, 64, wave * 16, 0, lane);
  const bf16x16 aq1 = frag_row(Qt, 64, wave * 16, 32, lane);
  floatx8 acc[4];
#pragma unroll
  for (int t = 0; t < 4; ++t) acc[t] = zero8();
  float zacc = 0.f;
  const float* ksrow = ksum + (size_t)nh * Mm;

  for (int m0 = 0; m0 < Mm; m0 += 32) {
    const int mt = m0 >> 4;
    // Qf tile: 16 rows x 32 m features
    const bf16x16 b0 = frag_packed(Ps, mt * 2, lane);
    const bf16x16 b1 = frag_packed(Ps, mt * 2 + 1, lane);
    const bf16x16 b2 = frag_packed(Ps, (mt + 1) * 2, lane);
    const bf16x16 b3 = frag_packed(Ps, (mt + 1) * 2 + 1, lane);
    floatx8 c0 = zero8();
    c0 = wmma_bf16(aq0, b0, c0);
    c0 = wmma_bf16(aq1, b1, c0);
    floatx8 c1 = zero8();
    c1 = wmma_bf16(aq0, b2, c1);
    c1 = wmma_bf16(aq1, b3, c1);
#pragma unroll
    for (int r = 0; r < 8; ++r) {
      const int row = r + 8 * half;
      const float sv = sh[wave * 16 + row];
      Qf[wave][row * 32 + (lane & 15)] =
          (__bf16)(RATIO * __expf(c0[r] - sv) + RKEPS);
      Qf[wave][row * 32 + 16 + (lane & 15)] =
          (__bf16)(RATIO * __expf(c1[r] - sv) + RKEPS);
    }
    // wave-private LDS; DS ops are in-order per wave -> no block barrier.

    // z partial: lane covers row (lane&15), m sub-chunk (half*16..+16)
    {
      const int row = lane & 15;
#pragma unroll
      for (int j = 0; j < 16; ++j) {
        const int mm = half * 16 + j;
        zacc += (float)Qf[wave][row * 32 + mm] * ksrow[m0 + mm];
      }
    }
    // out_tile += Qf[16l x 32m] * KV^T[32m x 64e]  (KV pre-swizzled)
    const bf16x16 aqf = frag_row(Qf[wave], 32, 0, 0, lane);
    const int mchunk = m0 >> 5;
#pragma unroll
    for (int t = 0; t < 4; ++t) {
      const bf16x16 bkv =
          frag_packed(KVs, ((nh * 64) + mchunk) * 4 + t, lane);
      acc[t] = wmma_bf16(aqf, bkv, acc[t]);
    }
  }

  zacc += __shfl_xor(zacc, 16, 32);  // combine the two m-half partials
  const float zv = 1.0f / (zacc + 1.0e-6f);  // lane L holds z for row L&15

#pragma unroll
  for (int t = 0; t < 4; ++t)
#pragma unroll
    for (int r = 0; r < 8; ++r) {
      const int row = r + 8 * half;
      const float zr = __shfl(zv, row, 32);  // lane 'row' holds z[row]
      const int l = l0 + wave * 16 + row;
      const int e = t * 16 + (lane & 15);
      out[((size_t)(n_i * Ln + l) * Hh + h_i) * Ee + e] = acc[t][r] * zr;
    }
}

extern "C" void kernel_launch(void* const* d_in, const int* in_sizes, int n_in,
                              void* d_out, int out_size, void* d_ws,
                              size_t ws_size, hipStream_t stream) {
  (void)in_sizes; (void)n_in; (void)out_size; (void)ws_size;
  const float* q = (const float*)d_in[0];
  const float* k = (const float*)d_in[1];
  const float* v = (const float*)d_in[2];
  const float* p = (const float*)d_in[3];
  float* out = (float*)d_out;

  // Workspace carve-out (~4.7 MB):
  //   shift_q | shift_k | ksum (f32) | Ps (swizzled bf16 P) | KVs (swizzled)
  float* ws = (float*)d_ws;
  float* shift_q = ws;                        // NHt*Ln f32
  float* shift_k = shift_q + NHt * Ln;        // NHt*Ln f32
  float* ksum = shift_k + NHt * Ln;           // NHt*Mm f32
  __bf16* Ps = (__bf16*)(ksum + NHt * Mm);    // Mm*Ee bf16 (256 KB)
  __bf16* KVs = Ps + (size_t)Mm * Ee;         // NHt*Ee*Mm bf16 (4 MB)

  dim3 grid(NHt * (Ln / 128));
  dim3 block(256);
  k_prep_p<<<dim3(32), block, 0, stream>>>(p, Ps);
  k_rowshift<<<grid, block, 0, stream>>>(q, Ps, shift_q);
  k_rowshift<<<grid, block, 0, stream>>>(k, Ps, shift_k);
  k_kv<<<dim3(NHt * (Mm / 128)), block, 0, stream>>>(k, v, Ps, shift_k, ksum,
                                                     KVs);
  k_out<<<grid, block, 0, stream>>>(q, Ps, shift_q, ksum, KVs, out);
}
